// SpectralConv1D_42717744726868
// MI455X (gfx1250) — matrix operand order repaired and checked by reading, not compiled, not run on hardware
//
#include <hip/hip_runtime.h>
#include <hip/hip_bf16.h>

// Problem constants (match reference)
#define BB 32
#define CI 64
#define CO 64
#define LL 8192
#define MODES 32
#define NFCOL 64            // 32 cos cols + 32 (-sin) cols
#define ROWS (BB*CI)        // 2048 rows (forward M; also B*CO for inverse)
#define KCHUNKS 8
#define KCHUNK (LL/KCHUNKS) // 1024

typedef __attribute__((ext_vector_type(16))) _Float16 v16h;
typedef __attribute__((ext_vector_type(8)))  _Float16 v8h;
typedef __attribute__((ext_vector_type(4)))  _Float16 v4h;
typedef __attribute__((ext_vector_type(8)))  float    v8f;
typedef __attribute__((ext_vector_type(4)))  float    v4f;

// ---------------------------------------------------------------------------
// Twiddle tables (2 MB total, L2-hot).
//   FT[j][n]  (64 x 8192): j<32 -> cos(2*pi*j*n/L), j>=32 -> -sin(...)
//   GT[n][k]  (8192 x 64): k<32 -> cos(2*pi*k*n/L), k>=32 -> +sin(...)
// (k*n) mod 8192 is exact in integer, so only f16 storage rounds.
// ---------------------------------------------------------------------------
__global__ __launch_bounds__(256) void twiddle_kernel(_Float16* __restrict__ FT,
                                                      _Float16* __restrict__ GT) {
    const int t = blockIdx.x * blockDim.x + threadIdx.x;   // 0 .. 64*8192-1
    if (t >= NFCOL * LL) return;
    const float W = 6.28318530717958647692f / (float)LL;
    {
        const int j = t >> 13;
        const int n = t & (LL - 1);
        const int m = ((j & 31) * n) & (LL - 1);
        float s, c;
        __sincosf((float)m * W, &s, &c);
        FT[t] = (_Float16)((j < 32) ? c : -s);
    }
    {
        const int n  = t >> 6;
        const int k2 = t & 63;
        const int m  = ((k2 & 31) * n) & (LL - 1);
        float s, c;
        __sincosf((float)m * W, &s, &c);
        GT[t] = (_Float16)((k2 < 32) ? c : s);
    }
}

// ---------------------------------------------------------------------------
// Forward thin DFT, split-K:
//   Xpart[chunk][2048][64] += x[2048][k0:k0+1024] * F[k0:k0+1024][64]
// Grid (128 M-tiles, 8 K-chunks), 4 waves/block; wave w owns N-tile w.
// Double-buffered LDS A-tile (16x32 f16), one barrier per K-step.
// ---------------------------------------------------------------------------
__global__ __launch_bounds__(128) void fwd_dft_kernel(const float* __restrict__ x,
                                                      const _Float16* __restrict__ FT,
                                                      float* __restrict__ Xpart) {
    __shared__ __align__(16) _Float16 As[2][16 * 32];

    const int mTile = blockIdx.x;              // 0..127
    const int chunk = blockIdx.y;              // 0..7
    const int k0    = chunk * KCHUNK;
    const int t     = threadIdx.x;
    const int wave  = t >> 5;                  // 0..3 -> N-tile
    const int lane  = t & 31;
    const int hsel  = lane >> 4;               // lane half (0/1)
    const int lm    = lane & 15;

    // cooperative A-load mapping: 128 threads * float4 = 16 rows x 32 cols
    const int arow = t >> 3;
    const int acol = (t << 2) & 31;
    const float* __restrict__ xrow =
        x + (size_t)(mTile * 16 + arow) * LL + k0;

    // FT is [64][8192]; lane reads 16 contiguous halves per K-step
    const _Float16* __restrict__ bp =
        FT + (size_t)(wave * 16 + lm) * LL + k0 + hsel * 16;

    // preload K-step 0 into buffer 0
    {
        v4f xv = *(const v4f*)(xrow + acol);
        v4h hx;
        hx[0] = (_Float16)xv[0]; hx[1] = (_Float16)xv[1];
        hx[2] = (_Float16)xv[2]; hx[3] = (_Float16)xv[3];
        *(v4h*)(&As[0][arow * 32 + acol]) = hx;
    }
    __syncthreads();

    v8f acc = {};
    const int NIT = KCHUNK / 32;               // 32 iterations

    for (int it = 0; it < NIT; ++it) {
        const int kk = it * 32;
        if (it + 1 < NIT) {
            if (k0 + kk + 288 <= LL)           // stream ahead (global_prefetch_b8)
                __builtin_prefetch(xrow + kk + 256, 0, 1);
            // prefetch next tile into the other LDS buffer
            v4f xv = *(const v4f*)(xrow + kk + 32 + acol);
            v4h hx;
            hx[0] = (_Float16)xv[0]; hx[1] = (_Float16)xv[1];
            hx[2] = (_Float16)xv[2]; hx[3] = (_Float16)xv[3];
            *(v4h*)(&As[(it + 1) & 1][arow * 32 + acol]) = hx;
        }

        // A fragment: lane K-run [h*16..+15] (contiguous, matches B)
        const _Float16* asrc = &As[it & 1][lm * 32 + hsel * 16];
        union { v16h v; v8h h[2]; } a;
        a.h[0] = *(const v8h*)(asrc);
        a.h[1] = *(const v8h*)(asrc + 8);

        // B fragment: lane halves are K = hsel*16 + 0..15 of column (wave*16+lm)
        v16h b = *(const v16h*)(bp + kk);

        acc = __builtin_amdgcn_wmma_f32_16x16x32_f16(
                  false, a.v, false, b, (short)0, acc, false, false);
        __syncthreads();
    }

    // C/D layout: lane -> col = lm, rows = hsel*8 + r
    float* __restrict__ Xc = Xpart + (size_t)chunk * (ROWS * NFCOL);
    const int row0 = mTile * 16 + hsel * 8;
    const int col  = wave * 16 + lm;
#pragma unroll
    for (int r = 0; r < 8; ++r)
        Xc[(size_t)(row0 + r) * NFCOL + col] = acc[r];
}

// ---------------------------------------------------------------------------
// Deterministic split-K reduction: X = sum_c Xpart[c]
// ---------------------------------------------------------------------------
__global__ __launch_bounds__(256) void reduce_kernel(const float* __restrict__ Xpart,
                                                     float* __restrict__ X) {
    const int idx = blockIdx.x * blockDim.x + threadIdx.x;
    if (idx >= ROWS * NFCOL) return;
    float s = 0.f;
#pragma unroll
    for (int c = 0; c < KCHUNKS; ++c)
        s += Xpart[(size_t)c * (ROWS * NFCOL) + idx];
    X[idx] = s;
}

// ---------------------------------------------------------------------------
// Mode mixing: Y[b,o,k] = sum_i X[b,i,k] * W[i,o,k]  (complex), with irfft
// constants folded into the f16 A-matrix of the inverse GEMM:
//   Yh[row][k] = c_k*Yr ; Yh[row][32+k] = -c_k*Yi   (c_0=1/L, c_k=2/L)
// ---------------------------------------------------------------------------
__global__ __launch_bounds__(256) void mode_mix_kernel(const float* __restrict__ X,
                                                       const float* __restrict__ wr,
                                                       const float* __restrict__ wi,
                                                       _Float16* __restrict__ Yh) {
    const int t = blockIdx.x * blockDim.x + threadIdx.x;   // 0..65535
    if (t >= BB * CO * MODES) return;
    const int b = t >> 11;
    const int o = (t >> 5) & 63;
    const int k = t & 31;

    float yr = 0.f, yi = 0.f;
#pragma unroll 4
    for (int i = 0; i < CI; ++i) {
        const float xr = X[(size_t)((b << 6) + i) * NFCOL + k];
        const float xi = X[(size_t)((b << 6) + i) * NFCOL + 32 + k];
        const float a  = wr[(size_t)((i << 6) + o) * MODES + k];
        const float bq = wi[(size_t)((i << 6) + o) * MODES + k];
        yr += xr * a - xi * bq;
        yi += xr * bq + xi * a;
    }
    const float ck = (k == 0 ? 1.0f : 2.0f) * (1.0f / (float)LL);
    _Float16* row = Yh + (size_t)((b << 6) + o) * NFCOL;
    row[k]      = (_Float16)(ck * yr);
    row[32 + k] = (_Float16)(-ck * yi);
}

// ---------------------------------------------------------------------------
// Inverse thin DFT:  out[2048][8192] = Yh[2048][64] * G[64][8192]
// K = 64 -> two WMMAs per 16x16 tile; 16384 blocks of 4 waves.
// ---------------------------------------------------------------------------
__global__ __launch_bounds__(128) void inv_dft_kernel(const _Float16* __restrict__ Yh,
                                                      const _Float16* __restrict__ GT,
                                                      float* __restrict__ out) {
    const int mTile  = blockIdx.y;                  // 0..127
    const int wave   = threadIdx.x >> 5;            // 0..3
    const int lane   = threadIdx.x & 31;
    const int hsel   = lane >> 4;
    const int lm     = lane & 15;
    const int nTile  = blockIdx.x * 4 + wave;       // 0..511

    const _Float16* __restrict__ ap = Yh + (size_t)(mTile * 16 + lm) * NFCOL;
    const _Float16* __restrict__ bp = GT + (size_t)(nTile * 16 + lm) * NFCOL
                                         + hsel * 16;
    v8f acc = {};
#pragma unroll
    for (int kk = 0; kk < NFCOL; kk += 32) {
        union { v16h v; v8h h[2]; } a;
        a.h[0] = *(const v8h*)(ap + kk + hsel * 16);
        a.h[1] = *(const v8h*)(ap + kk + hsel * 16 + 8);
        v16h b = *(const v16h*)(bp + kk);
        acc = __builtin_amdgcn_wmma_f32_16x16x32_f16(
                  false, a.v, false, b, (short)0, acc, false, false);
    }

    const int row0 = mTile * 16 + hsel * 8;
    const int col  = nTile * 16 + lm;
#pragma unroll
    for (int r = 0; r < 8; ++r)
        out[(size_t)(row0 + r) * LL + col] = acc[r];
}

// ---------------------------------------------------------------------------
extern "C" void kernel_launch(void* const* d_in, const int* in_sizes, int n_in,
                              void* d_out, int out_size, void* d_ws, size_t ws_size,
                              hipStream_t stream) {
    const float* x  = (const float*)d_in[0];   // [32,64,8192]
    const float* wr = (const float*)d_in[1];   // [64,64,32]
    const float* wi = (const float*)d_in[2];   // [64,64,32]
    float* out = (float*)d_out;                // [32,64,8192]

    // workspace layout
    char* ws = (char*)d_ws;
    _Float16* FT    = (_Float16*)(ws);                       // 1 MB
    _Float16* GT    = (_Float16*)(ws + (1u << 20));          // 1 MB
    float*    Xpart = (float*)   (ws + (2u << 20));          // 8 * 512 KB = 4 MB
    float*    X     = (float*)   (ws + (6u << 20));          // 512 KB
    _Float16* Yh    = (_Float16*)(ws + (6u << 20) + (512u << 10)); // 256 KB

    // 1) twiddle tables
    twiddle_kernel<<<(NFCOL * LL) / 256, 256, 0, stream>>>(FT, GT);

    // 2) forward thin DFT (WMMA GEMM, split-K for occupancy: 1024 blocks)
    fwd_dft_kernel<<<dim3(ROWS / 16, KCHUNKS), 128, 0, stream>>>(x, FT, Xpart);

    // 3) deterministic split-K reduction
    reduce_kernel<<<(ROWS * NFCOL) / 256, 256, 0, stream>>>(Xpart, X);

    // 4) complex per-mode channel mixing + irfft constant folding
    mode_mix_kernel<<<(BB * CO * MODES) / 256, 256, 0, stream>>>(X, wr, wi, Yh);

    // 5) inverse thin DFT (WMMA GEMM, K=64, N=8192)
    inv_dft_kernel<<<dim3(LL / 16 / 4, ROWS / 16), 128, 0, stream>>>(Yh, GT, out);
}
